// SigmaMoE_31439160607027
// MI455X (gfx1250) — matrix operand order repaired and checked by reading, not compiled, not run on hardware
//
#include <hip/hip_runtime.h>
#include <hip/hip_bf16.h>

typedef __attribute__((ext_vector_type(16))) __bf16 bf16x16;
typedef __attribute__((ext_vector_type(8)))  __bf16 bf16x8;
typedef __attribute__((ext_vector_type(8)))  float  f32x8;

#define T_TOK   8192
#define DIM     512
#define NEXP    32
#define HID     128
#define CAP     8192           // max tokens per expert (<= T)
#define WAVES   4              // waves per block in the GEMM kernel
#define HSTRIDE 136            // LDS row stride in bf16 (16B-aligned, conflict-padded)

// ---------------------------------------------------------------- zero output
__global__ void moe_zero_out(float* __restrict__ out, int n) {
    int i = blockIdx.x * blockDim.x + threadIdx.x;
    if (i < n) out[i] = 0.0f;
}

// ---------------------------------------------------------------- router
// one wave per token; lane = expert. dot(x[t], expert_sel[e]) -> sigmoid -> top2
__global__ void moe_router(const float* __restrict__ x,
                           const float* __restrict__ esel,
                           int* __restrict__ top2_idx,
                           float* __restrict__ top2_w) {
    int lane = threadIdx.x & 31;
    int wv   = threadIdx.x >> 5;
    int t    = blockIdx.x * (blockDim.x >> 5) + wv;
    if (t >= T_TOK) return;

    const float* xr = x + (size_t)t * DIM;
    const float* sr = esel + (size_t)lane * DIM;
    float dot = 0.0f;
    #pragma unroll 4
    for (int k = 0; k < DIM; k += 4) {
        float4 xv = *(const float4*)(xr + k);
        float4 sv = *(const float4*)(sr + k);
        dot = fmaf(xv.x, sv.x, dot);
        dot = fmaf(xv.y, sv.y, dot);
        dot = fmaf(xv.z, sv.z, dot);
        dot = fmaf(xv.w, sv.w, dot);
    }
    float s = 1.0f / (1.0f + __expf(-dot));

    // top-1
    float v = s; int id = lane;
    #pragma unroll
    for (int off = 16; off >= 1; off >>= 1) {
        float ov = __shfl_xor(v, off, 32);
        int   oi = __shfl_xor(id, off, 32);
        if (ov > v || (ov == v && oi < id)) { v = ov; id = oi; }
    }
    int idx0 = id; float w0 = v;
    // top-2
    float s2 = (lane == idx0) ? -1e30f : s;
    v = s2; id = lane;
    #pragma unroll
    for (int off = 16; off >= 1; off >>= 1) {
        float ov = __shfl_xor(v, off, 32);
        int   oi = __shfl_xor(id, off, 32);
        if (ov > v || (ov == v && oi < id)) { v = ov; id = oi; }
    }
    int idx1 = id; float w1 = v;

    if (lane == 0) {
        top2_idx[2 * t]     = idx0;
        top2_idx[2 * t + 1] = idx1;
        top2_w[2 * t]       = w0;
        top2_w[2 * t + 1]   = w1;
    }
}

// ---------------------------------------------------------------- scatter
// one wave per expert: stable ballot-compaction of matching tokens
__global__ void moe_scatter(const int* __restrict__ top2_idx,
                            const float* __restrict__ top2_w,
                            int* __restrict__ token_list,
                            float* __restrict__ weight_list,
                            int* __restrict__ counts) {
    int lane = threadIdx.x & 31;
    int e    = threadIdx.x >> 5;          // one block of 1024 threads = 32 waves
    if (e >= NEXP) return;
    int cnt = 0;
    for (int base = 0; base < T_TOK; base += 32) {
        int t = base + lane;
        int i0 = top2_idx[2 * t];
        int i1 = top2_idx[2 * t + 1];
        bool m = (i0 == e) || (i1 == e);
        float w = (i0 == e) ? top2_w[2 * t] : top2_w[2 * t + 1];
        unsigned bal = (unsigned)__ballot(m);
        int pre = __popc(bal & ((1u << lane) - 1u));
        if (m) {
            token_list[e * CAP + cnt + pre]  = t;
            weight_list[e * CAP + cnt + pre] = w;
        }
        cnt += __popc(bal);
    }
    if (lane == 0) counts[e] = cnt;
}

// ---------------------------------------------------------------- pack x -> bf16
__global__ void moe_pack_x(const float* __restrict__ x, __bf16* __restrict__ xb, int n) {
    int i = blockIdx.x * blockDim.x + threadIdx.x;
    if (i < n) xb[i] = (__bf16)x[i];
}

// ---------------------------------------------------------------- pack keys
// dst[(((e*8+j)*16+ks)*32+lane)*16+elem], k = ks*32 + 8*half + (elem&7) + 16*(elem>>3), h = j*16 + n
__global__ void moe_pack_keys(const float* __restrict__ keys, __bf16* __restrict__ dst) {
    int tid  = blockIdx.x * blockDim.x + threadIdx.x;
    int elem = tid & 15;
    int lane = (tid >> 4) & 31;
    int ks   = (tid >> 9) & 15;
    int j    = (tid >> 13) & 7;
    int e    = tid >> 16;
    if (e >= NEXP) return;
    int n    = lane & 15, half = lane >> 4;
    int k    = ks * 32 + half * 8 + (elem & 7) + ((elem >> 3) << 4);
    int h    = j * 16 + n;
    dst[tid] = (__bf16)keys[((size_t)e * DIM + k) * HID + h];
}

// ---------------------------------------------------------------- pack values
// dst[(((e*32+v)*4+q)*32+lane)*16+elem], k = q*32 + 8*half + (elem&7) + 16*(elem>>3), col = v*16 + n
__global__ void moe_pack_values(const float* __restrict__ values, __bf16* __restrict__ dst) {
    int tid  = blockIdx.x * blockDim.x + threadIdx.x;
    int elem = tid & 15;
    int lane = (tid >> 4) & 31;
    int q    = (tid >> 9) & 3;
    int v    = (tid >> 11) & 31;
    int e    = tid >> 16;
    if (e >= NEXP) return;
    int n    = lane & 15, half = lane >> 4;
    int k    = q * 32 + half * 8 + (elem & 7) + ((elem >> 3) << 4);
    int col  = v * 16 + n;
    dst[tid] = (__bf16)values[((size_t)e * HID + k) * DIM + col];
}

// ---------------------------------------------------------------- grouped GEMM
// one wave handles one 32-token tile (two 16-row M-tiles) of one expert:
//   h = relu(Xg[32x512] @ keys[e][512x128]) * w   (phase 1, 256 WMMAs)
//   out[tokens] += h @ values[e][128x512]          (phase 2, 256 WMMAs + atomics)
// B fragments are shared by both M-tiles (2x reuse) and software-pipelined.
__global__ void __launch_bounds__(WAVES * 32)
moe_gemm(const __bf16* __restrict__ xb,
         const __bf16* __restrict__ kpk,
         const __bf16* __restrict__ vpk,
         const int* __restrict__ token_list,
         const float* __restrict__ weight_list,
         const int* __restrict__ counts,
         float* __restrict__ out) {
    __shared__ __bf16 hsh[WAVES][32 * HSTRIDE];

    int lane = threadIdx.x & 31;
    int wv   = threadIdx.x >> 5;
    int e    = blockIdx.y;
    int tile = blockIdx.x * WAVES + wv;
    int cnt  = counts[e];
    if (tile * 32 >= cnt) return;

    int n = lane & 15, half = lane >> 4;
    int base = tile * 32;

    // tokens/weights for rows n (M-tile 0) and 16+n (M-tile 1); padded rows -> w=0
    int t0 = 0, t1 = 0; float w0 = 0.0f, w1 = 0.0f;
    {
        int i0 = base + n;
        int i1 = base + 16 + n;
        if (i0 < cnt) { t0 = token_list[e * CAP + i0]; w0 = weight_list[e * CAP + i0]; }
        if (i1 < cnt) { t1 = token_list[e * CAP + i1]; w1 = weight_list[e * CAP + i1]; }
    }
    // per-fragment-row token/weight (rows r + 8*half within each M-tile)
    int tr[2][8]; float wr[2][8];
    #pragma unroll
    for (int r = 0; r < 8; r++) {
        int src = r + 8 * half;
        tr[0][r] = __shfl(t0, src, 32); wr[0][r] = __shfl(w0, src, 32);
        tr[1][r] = __shfl(t1, src, 32); wr[1][r] = __shfl(w1, src, 32);
    }

    const __bf16* xr0 = xb + (size_t)t0 * DIM;
    const __bf16* xr1 = xb + (size_t)t1 * DIM;

    auto loadA = [&](const __bf16* xr, int ks) -> bf16x16 {
        int k0 = ks * 32 + half * 8;
        bf16x8 a0 = *(const bf16x8*)(xr + k0);
        bf16x8 a1 = *(const bf16x8*)(xr + k0 + 16);
        return __builtin_shufflevector(a0, a1, 0,1,2,3,4,5,6,7,8,9,10,11,12,13,14,15);
    };
    auto loadBK = [&](int j, int ks) -> bf16x16 {
        const __bf16* bp = kpk + ((((size_t)e * 8 + j) * 16 + ks) * 32 + lane) * 16;
        bf16x8 b0 = *(const bf16x8*)(bp);
        bf16x8 b1 = *(const bf16x8*)(bp + 8);
        return __builtin_shufflevector(b0, b1, 0,1,2,3,4,5,6,7,8,9,10,11,12,13,14,15);
    };
    auto loadBV = [&](int v, int q) -> bf16x16 {
        const __bf16* bp = vpk + ((((size_t)e * 32 + v) * 4 + q) * 32 + lane) * 16;
        bf16x8 b0 = *(const bf16x8*)(bp);
        bf16x8 b1 = *(const bf16x8*)(bp + 8);
        return __builtin_shufflevector(b0, b1, 0,1,2,3,4,5,6,7,8,9,10,11,12,13,14,15);
    };

    // ---------------- phase 1: h = X @ keys[e], software-pipelined
    f32x8 zero = {};
    f32x8 acc[2][8];
    #pragma unroll
    for (int j = 0; j < 8; j++) { acc[0][j] = zero; acc[1][j] = zero; }

    bf16x16 B0[4], B1[4];
    #pragma unroll
    for (int j = 0; j < 4; j++) B0[j] = loadBK(j, 0);
    bf16x16 A0 = loadA(xr0, 0);
    bf16x16 A1 = loadA(xr1, 0);

    for (int ks = 0; ks < 16; ks++) {
        // fetch group1 fragments for this k-step (overlaps with group0 WMMAs)
        #pragma unroll
        for (int j = 0; j < 4; j++) B1[j] = loadBK(4 + j, ks);
        #pragma unroll
        for (int j = 0; j < 4; j++) {
            acc[0][j] = __builtin_amdgcn_wmma_f32_16x16x32_bf16(
                false, A0, false, B0[j], (short)0, acc[0][j], false, false);
            acc[1][j] = __builtin_amdgcn_wmma_f32_16x16x32_bf16(
                false, A1, false, B0[j], (short)0, acc[1][j], false, false);
        }
        if (ks < 15) {
            #pragma unroll
            for (int j = 0; j < 4; j++) B0[j] = loadBK(j, ks + 1);
        }
        #pragma unroll
        for (int j = 0; j < 4; j++) {
            acc[0][4 + j] = __builtin_amdgcn_wmma_f32_16x16x32_bf16(
                false, A0, false, B1[j], (short)0, acc[0][4 + j], false, false);
            acc[1][4 + j] = __builtin_amdgcn_wmma_f32_16x16x32_bf16(
                false, A1, false, B1[j], (short)0, acc[1][4 + j], false, false);
        }
        if (ks < 15) {
            A0 = loadA(xr0, ks + 1);
            A1 = loadA(xr1, ks + 1);
        }
    }

    // relu + fold routing weight, stage h (32x128 bf16) in per-wave LDS
    #pragma unroll
    for (int mt = 0; mt < 2; mt++) {
        #pragma unroll
        for (int j = 0; j < 8; j++) {
            #pragma unroll
            for (int r = 0; r < 8; r++) {
                int m = mt * 16 + r + 8 * half;
                float v = fmaxf(acc[mt][j][r], 0.0f) * wr[mt][r];
                hsh[wv][m * HSTRIDE + j * 16 + n] = (__bf16)v;
            }
        }
    }

    // phase-2 A fragments from LDS (rows lane&15 and 16+(lane&15))
    bf16x16 A2[2][4];
    #pragma unroll
    for (int mt = 0; mt < 2; mt++) {
        const __bf16* hr = &hsh[wv][(mt * 16 + (lane & 15)) * HSTRIDE];
        #pragma unroll
        for (int q = 0; q < 4; q++) {
            int k0 = q * 32 + half * 8;
            bf16x8 a0 = *(const bf16x8*)(hr + k0);
            bf16x8 a1 = *(const bf16x8*)(hr + k0 + 16);
            A2[mt][q] = __builtin_shufflevector(a0, a1,
                0,1,2,3,4,5,6,7,8,9,10,11,12,13,14,15);
        }
    }

    // ---------------- phase 2: out += h @ values[e], double-buffered over v
    bf16x16 Bb[2][4];
    #pragma unroll
    for (int q = 0; q < 4; q++) Bb[0][q] = loadBV(0, q);

    #pragma unroll 2
    for (int v = 0; v < 32; v++) {
        int cur = v & 1, nxt = cur ^ 1;
        if (v < 31) {
            #pragma unroll
            for (int q = 0; q < 4; q++) Bb[nxt][q] = loadBV(v + 1, q);
        }
        f32x8 c0 = zero, c1 = zero;
        #pragma unroll
        for (int q = 0; q < 4; q++) {
            c0 = __builtin_amdgcn_wmma_f32_16x16x32_bf16(
                false, A2[0][q], false, Bb[cur][q], (short)0, c0, false, false);
            c1 = __builtin_amdgcn_wmma_f32_16x16x32_bf16(
                false, A2[1][q], false, Bb[cur][q], (short)0, c1, false, false);
        }
        int col = v * 16 + n;
        #pragma unroll
        for (int r = 0; r < 8; r++) {
            unsafeAtomicAdd(out + (size_t)tr[0][r] * DIM + col, c0[r]);
            unsafeAtomicAdd(out + (size_t)tr[1][r] * DIM + col, c1[r]);
        }
    }
}

// ---------------------------------------------------------------- launch
extern "C" void kernel_launch(void* const* d_in, const int* in_sizes, int n_in,
                              void* d_out, int out_size, void* d_ws, size_t ws_size,
                              hipStream_t stream) {
    const float* x      = (const float*)d_in[0];   // [T, 512]
    const float* esel   = (const float*)d_in[1];   // [32, 512]
    const float* keys   = (const float*)d_in[2];   // [32, 512, 128]
    const float* values = (const float*)d_in[3];   // [32, 128, 512]
    float* out = (float*)d_out;

    char* ws = (char*)d_ws;
    size_t off = 0;
    auto alloc = [&](size_t bytes) -> void* {
        void* p = ws + off;
        off = (off + bytes + 255) & ~(size_t)255;
        return p;
    };
    int*    counts      = (int*)alloc(NEXP * sizeof(int));
    int*    top2_idx    = (int*)alloc((size_t)T_TOK * 2 * sizeof(int));
    float*  top2_w      = (float*)alloc((size_t)T_TOK * 2 * sizeof(float));
    int*    token_list  = (int*)alloc((size_t)NEXP * CAP * sizeof(int));
    float*  weight_list = (float*)alloc((size_t)NEXP * CAP * sizeof(float));
    __bf16* xb          = (__bf16*)alloc((size_t)T_TOK * DIM * sizeof(__bf16));
    __bf16* kpk         = (__bf16*)alloc((size_t)NEXP * DIM * HID * sizeof(__bf16));
    __bf16* vpk         = (__bf16*)alloc((size_t)NEXP * DIM * HID * sizeof(__bf16));

    // zero the output (atomically accumulated later)
    moe_zero_out<<<(out_size + 255) / 256, 256, 0, stream>>>(out, out_size);

    // router: 8 waves / block, one wave per token
    moe_router<<<T_TOK / 8, 256, 0, stream>>>(x, esel, top2_idx, top2_w);

    // scatter: one wave per expert (32 waves in one block)
    moe_scatter<<<1, 1024, 0, stream>>>(top2_idx, top2_w, token_list, weight_list, counts);

    // precision/layout packing
    {
        int n = T_TOK * DIM;
        moe_pack_x<<<(n + 255) / 256, 256, 0, stream>>>(x, xb, n);
    }
    {
        int n = NEXP * DIM * HID;   // 2,097,152
        moe_pack_keys<<<(n + 255) / 256, 256, 0, stream>>>(keys, kpk);
        moe_pack_values<<<(n + 255) / 256, 256, 0, stream>>>(values, vpk);
    }

    // grouped GEMM: grid.y = expert, grid.x covers up to 256 tiles/expert (32 tok each)
    dim3 grid((CAP / 32 + WAVES - 1) / WAVES, NEXP);
    moe_gemm<<<grid, WAVES * 32, 0, stream>>>(xb, kpk, vpk, token_list, weight_list,
                                              counts, out);
}
